// ProbAttention_67619965108933
// MI455X (gfx1250) — matrix operand order repaired and checked by reading, not compiled, not run on hardware
//
#include <hip/hip_runtime.h>
#include <hip/hip_fp16.h>
#include <math.h>

typedef __attribute__((ext_vector_type(16))) _Float16 v16h;
typedef __attribute__((ext_vector_type(8)))  float    v8f;

#define B_   2
#define L_   4096
#define H_   8
#define D_   64
#define SK_  45       // sample_k
#define NT_  45       // n_top
#define SCALE_ 0.125f // 1/sqrt(64)

// ---------------------------------------------------------------- kernel 0
__global__ void pa_zero_out(float* __restrict__ out, int n4) {
  int i = blockIdx.x * blockDim.x + threadIdx.x;
  if (i < n4) ((float4*)out)[i] = make_float4(0.f, 0.f, 0.f, 0.f);
}

// ---------------------------------------------------------------- kernel 1
// M[b,h,l] = max_s(q·k_sample) - sum_s(q·k_sample)/L   (one wave per query)
__global__ void pa_compute_M(const float* __restrict__ q,
                             const float* __restrict__ k,
                             const long long* __restrict__ idxs,
                             float* __restrict__ Mout) {
  int wave = blockIdx.x * (blockDim.x >> 5) + (threadIdx.x >> 5);
  int lane = threadIdx.x & 31;
  int l  = wave & (L_ - 1);
  int hb = wave >> 12;             // b*H + h
  int h  = hb & (H_ - 1);
  int b  = hb >> 3;
  const float* qrow = q + ((((long)b * L_ + l) * H_ + h) << 6);
  float q0 = qrow[lane], q1 = qrow[lane + 32];
  float mx = -INFINITY, sm = 0.f;
  for (int s = 0; s < SK_; ++s) {
    int ki = (int)idxs[(long)l * SK_ + s];
    const float* krow = k + ((((long)b * L_ + ki) * H_ + h) << 6);
    float d = q0 * krow[lane] + q1 * krow[lane + 32];
#pragma unroll
    for (int m = 16; m >= 1; m >>= 1) d += __shfl_xor(d, m, 32);
    mx = fmaxf(mx, d);
    sm += d;
  }
  if (lane == 0) Mout[((long)hb << 12) + l] = mx - sm * (1.0f / L_);
}

// ---------------------------------------------------------------- kernel 2
// exact top-45 selection per (b,h); M values LDS-resident (16 KB)
__global__ void pa_topk(const float* __restrict__ Min, int* __restrict__ Mtop) {
  __shared__ float vals[L_];
  __shared__ float rv[256];
  __shared__ int   ri[256];
  int bh = blockIdx.x, t = threadIdx.x;
  for (int j = t; j < L_; j += 256) vals[j] = Min[(long)bh * L_ + j];
  __syncthreads();
  for (int sel = 0; sel < NT_; ++sel) {
    float best = -INFINITY; int bi = L_;
#pragma unroll
    for (int j = 0; j < 16; ++j) {
      int idx = t * 16 + j;
      float v = vals[idx];
      if (v > best || (v == best && idx < bi)) { best = v; bi = idx; }
    }
    rv[t] = best; ri[t] = bi;
    __syncthreads();
    for (int s = 128; s >= 1; s >>= 1) {
      if (t < s) {
        float ov = rv[t + s]; int oi = ri[t + s];
        if (ov > rv[t] || (ov == rv[t] && oi < ri[t])) { rv[t] = ov; ri[t] = oi; }
      }
      __syncthreads();
    }
    if (t == 0) { Mtop[bh * NT_ + sel] = ri[0]; vals[ri[0]] = -INFINITY; }
    __syncthreads();
  }
}

// ---------------------------------------------------------------- kernel 3
// A-operand (16x32 f16) k order chosen here (B uses the same): lane holds row m=lane&15; halves e map to
// K: k=e+16g  (g=lane>>4)
__device__ __forceinline__ int a_kmap(int e, int g) {
  return e + 16 * g;
}

__global__ void __launch_bounds__(256)
pa_attention(const float* __restrict__ q,
             const float* __restrict__ kk,
             const float* __restrict__ vv,
             const int* __restrict__ Mtop,
             float* __restrict__ out) {
  const int KW = 8;                       // waves per block, split over keys
  __shared__ _Float16 Pbuf[KW][16][32];   // C-layout -> A-layout staging
  __shared__ float cm[KW][16], cl[KW][16];
  __shared__ float cO[KW][16][64];

  int mtile = blockIdx.x % 3;             // 48-padded query rows, 16 per tile
  int bh = blockIdx.x / 3;
  int h = bh & 7, b = bh >> 3;
  int wave = threadIdx.x >> 5, lane = threadIdx.x & 31;
  int g = lane >> 4, mr = lane & 15;

  // ---- load Q_reduce tile (scaled) as two A matrices: d=0..31 and 32..63
  v16h aq0, aq1;
  {
    int u = mtile * 16 + mr;
    const float* qrow = nullptr;
    if (u < NT_) {
      int qi = Mtop[bh * NT_ + u];
      qrow = q + ((((long)b * L_ + qi) * H_ + h) << 6);
    }
#pragma unroll
    for (int e = 0; e < 16; ++e) {
      int d = a_kmap(e, g);
      aq0[e] = (_Float16)(qrow ? qrow[d]      * SCALE_ : 0.f);
      aq1[e] = (_Float16)(qrow ? qrow[d + 32] * SCALE_ : 0.f);
    }
  }

  float mrow[8], lrow[8];
  v8f O[4] = {v8f{}, v8f{}, v8f{}, v8f{}};
#pragma unroll
  for (int r = 0; r < 8; ++r) { mrow[r] = -INFINITY; lrow[r] = 0.f; }

  const int keys_per_wave = L_ / KW;      // 512
  const int kstart = wave * keys_per_wave;

  for (int kb = kstart; kb < kstart + keys_per_wave; kb += 32) {
    if (kb + 32 < kstart + keys_per_wave) {
      const float* pf = kk + ((((long)b * L_ + (kb + 32 + lane)) * H_ + h) << 6);
      __builtin_prefetch(pf, 0, 0);       // global_prefetch next key tile
    }
    // ---- S = (Q*scale) @ K^T for 32 keys: two 16-key C/D tiles
    v8f S[2];
#pragma unroll
    for (int sub = 0; sub < 2; ++sub) {
      int key = kb + sub * 16 + mr;       // B col n = key
      const float* krow = kk + ((((long)b * L_ + key) * H_ + h) << 6);
      v16h b0, b1;
#pragma unroll
      for (int e = 0; e < 16; ++e) {      // B row k = d = e + 16g
        int d = e + 16 * g;
        b0[e] = (_Float16)krow[d];
        b1[e] = (_Float16)krow[d + 32];
      }
      v8f c = {};
      c = __builtin_amdgcn_wmma_f32_16x16x32_f16(false, aq0, false, b0, (short)0, c, false, false);
      c = __builtin_amdgcn_wmma_f32_16x16x32_f16(false, aq1, false, b1, (short)0, c, false, false);
      S[sub] = c;
    }
    // ---- online softmax (row = r + 8g; 16 lanes of a half share a row)
    float p0[8], p1[8];
#pragma unroll
    for (int r = 0; r < 8; ++r) {
      float rx = fmaxf(S[0][r], S[1][r]);
#pragma unroll
      for (int msk = 8; msk >= 1; msk >>= 1) rx = fmaxf(rx, __shfl_xor(rx, msk, 32));
      float nm = fmaxf(mrow[r], rx);
      float corr = __expf(mrow[r] - nm);
      mrow[r] = nm;
      p0[r] = __expf(S[0][r] - nm);
      p1[r] = __expf(S[1][r] - nm);
      float rs = p0[r] + p1[r];
#pragma unroll
      for (int msk = 8; msk >= 1; msk >>= 1) rs += __shfl_xor(rs, msk, 32);
      lrow[r] = lrow[r] * corr + rs;
      O[0][r] *= corr; O[1][r] *= corr; O[2][r] *= corr; O[3][r] *= corr;
    }
    // ---- P: C-layout -> A-layout through per-wave LDS staging
#pragma unroll
    for (int r = 0; r < 8; ++r) {
      Pbuf[wave][r + 8 * g][mr]      = (_Float16)p0[r];
      Pbuf[wave][r + 8 * g][mr + 16] = (_Float16)p1[r];
    }
    asm volatile("s_wait_dscnt 0" ::: "memory");
    v16h ap;
#pragma unroll
    for (int e = 0; e < 16; ++e) ap[e] = Pbuf[wave][mr][a_kmap(e, g)];
    // ---- O += P @ V (V as 4 B-tiles of 32x16, d = 16t + n)
#pragma unroll
    for (int t = 0; t < 4; ++t) {
      v16h bv;
#pragma unroll
      for (int e = 0; e < 16; ++e) {
        int key = kb + e + 16 * g;
        bv[e] = (_Float16)vv[((((long)b * L_ + key) * H_ + h) << 6) + t * 16 + mr];
      }
      O[t] = __builtin_amdgcn_wmma_f32_16x16x32_f16(false, ap, false, bv, (short)0, O[t], false, false);
    }
  }

  // ---- publish per-wave partials
  if (mr == 0) {
#pragma unroll
    for (int r = 0; r < 8; ++r) { cm[wave][r + 8 * g] = mrow[r]; cl[wave][r + 8 * g] = lrow[r]; }
  }
#pragma unroll
  for (int r = 0; r < 8; ++r) {
    cO[wave][r + 8 * g][mr]      = O[0][r];
    cO[wave][r + 8 * g][mr + 16] = O[1][r];
    cO[wave][r + 8 * g][mr + 32] = O[2][r];
    cO[wave][r + 8 * g][mr + 48] = O[3][r];
  }
  __syncthreads();

  // ---- merge the 8 key-split partials, normalize, scatter to output
  for (int idx = threadIdx.x; idx < 16 * 64; idx += 256) {
    int row = idx >> 6, col = idx & 63;
    float gm = -INFINITY;
#pragma unroll
    for (int w = 0; w < KW; ++w) gm = fmaxf(gm, cm[w][row]);
    float den = 0.f, num = 0.f;
#pragma unroll
    for (int w = 0; w < KW; ++w) {
      float f = __expf(cm[w][row] - gm);
      den += cl[w][row] * f;
      num += cO[w][row][col] * f;
    }
    int u = mtile * 16 + row;
    if (u < NT_) {
      int oi = Mtop[bh * NT_ + u];
      out[((((long)b * L_ + oi) * H_ + h) << 6) + col] = num / den;
    }
  }
}

// ---------------------------------------------------------------- launcher
extern "C" void kernel_launch(void* const* d_in, const int* in_sizes, int n_in,
                              void* d_out, int out_size, void* d_ws, size_t ws_size,
                              hipStream_t stream) {
  const float*     q    = (const float*)d_in[0];
  const float*     k    = (const float*)d_in[1];
  const float*     v    = (const float*)d_in[2];
  const long long* idxs = (const long long*)d_in[4];   // index_sample, int64
  float* out = (float*)d_out;

  float* wsM   = (float*)d_ws;                                       // B*H*L floats
  int*   wsTop = (int*)((char*)d_ws + (size_t)B_ * H_ * L_ * sizeof(float));

  // zero the scatter target (reference context starts as zeros)
  pa_zero_out<<<(out_size / 4 + 255) / 256, 256, 0, stream>>>(out, out_size / 4);
  // sparsity scores M
  pa_compute_M<<<(B_ * H_ * L_) / 8, 256, 0, stream>>>(q, k, idxs, wsM);
  // exact top-45 per (b,h)
  pa_topk<<<B_ * H_, 256, 0, stream>>>(wsM, wsTop);
  // WMMA flash attention for the selected queries + scatter
  pa_attention<<<B_ * H_ * 3, 256, 0, stream>>>(q, k, v, wsTop, out);
}
